// VanillaLSTM_44839458570396
// MI455X (gfx1250) — matrix operand order repaired and checked by reading, not compiled, not run on hardware
//
#include <hip/hip_runtime.h>
#include <hip/hip_bf16.h>

// ---------------------------------------------------------------------------
// Persistent bf16-WMMA LSTM for gfx1250 (MI455X).
// Each WG owns 16 batch rows for ALL 256 timesteps (rows are independent).
// c stays in VGPRs, h ping-pongs between two LDS buffers (bf16). Weights are
// pre-swizzled into exact v_wmma_f32_16x16x32_bf16 B-fragment order; a single
// per-lane base pointer + 24-bit immediate offsets covers all 96 B loads.
// Per K-step, all 8 B fragments are loaded as one burst (16 global_load_b128
// in a clause) before the 8 WMMAs consume them with descending loadcnt waits.
// ---------------------------------------------------------------------------

typedef __attribute__((ext_vector_type(16))) __bf16          v16bf;
typedef __attribute__((ext_vector_type(16))) unsigned short  v16u;
typedef __attribute__((ext_vector_type(8)))  unsigned short  v8u;
typedef __attribute__((ext_vector_type(8)))  float           v8f;

#define TSTEPS 256
#define NROWS  16384
#define HDIM   256
#define MDIM   128
#define KSTEPS 12           // 384 / 32
#define RST    136          // smR row stride (bf16 elems), mult of 8, bank-skewed
#define HST    264          // smH row stride (bf16 elems), mult of 8, bank-skewed

__device__ __forceinline__ unsigned short f2bf(float x) {
  unsigned u = __builtin_bit_cast(unsigned, x);
  unsigned r = (u + 0x7FFFu + ((u >> 16) & 1u)) >> 16;   // round-nearest-even
  return (unsigned short)r;
}
__device__ __forceinline__ float bf2f(unsigned short s) {
  unsigned u = ((unsigned)s) << 16;
  return __builtin_bit_cast(float, u);
}

// native TRANS-unit math: v_exp_f32 / v_rcp_f32 (and v_tanh_f32 if exposed)
__device__ __forceinline__ float sigm(float x) {
  return __builtin_amdgcn_rcpf(1.f + __builtin_amdgcn_exp2f(x * -1.44269504f));
}
#if __has_builtin(__builtin_amdgcn_tanhf)
__device__ __forceinline__ float ftanh(float x) { return __builtin_amdgcn_tanhf(x); }
#else
__device__ __forceinline__ float ftanh(float x) {
  float e = __builtin_amdgcn_exp2f(x * 2.88539008f);     // exp(2x)
  return 1.f - 2.f * __builtin_amdgcn_rcpf(e + 1.f);
}
#endif

// B fragment (kk, j) with j = q*2+s: 32B per lane, base + compile-time imm
__device__ __forceinline__ v16bf loadB_frag(const unsigned char* bB, int kk, int j) {
  int q = j >> 1, s = j & 1;
  const v16u* bp = (const v16u*)(bB + (size_t)((kk * 64 + q * 16 + s) * 1024));
  return __builtin_bit_cast(v16bf, *bp);
}

// A fragment for K-step kk: lane row=lm, two 8-elem runs (2x ds_load_b128)
__device__ __forceinline__ v16bf loadA_frag(const unsigned short* smR_,
                                            const unsigned short* hbase,
                                            int lm, int g, int kk) {
  const unsigned short* ap = (kk < 4)
      ? &smR_[lm * RST + kk * 32 + g * 8]
      : &hbase[lm * HST + (kk - 4) * 32 + g * 8];
  v8u alo = *(const v8u*)ap;
  v8u ahi = *(const v8u*)(ap + 16);
  v16u au = __builtin_shufflevector(alo, ahi,
              0, 1, 2, 3, 4, 5, 6, 7, 8, 9, 10, 11, 12, 13, 14, 15);
  return __builtin_bit_cast(v16bf, au);
}

// --- prep: swizzle [K=384][1024] B-matrix (bf16) into per-lane fragment order
// B[k][col] = k<128 ? W_ih[col][k] : W_hh[col][k-128]
// idx = (((kk*64 + ctile)*32 + lane)*16 + e), K = kk*32 + (e/8)*16 + (lane/16)*8 + e%8,
// col = ctile*16 + lane%16   -> each lane's 16 elems are contiguous (32B)
__global__ __launch_bounds__(256) void lstm_prep_w(const float* __restrict__ W_ih,
                                                   const float* __restrict__ W_hh,
                                                   unsigned short* __restrict__ Wsw) {
  int idx = blockIdx.x * 256 + threadIdx.x;      // 0 .. 393215
  int e   = idx & 15;
  int L   = (idx >> 4) & 31;
  int c   = (idx >> 9) & 63;
  int kk  = idx >> 15;
  int k   = kk * 32 + (e >> 3) * 16 + (L >> 4) * 8 + (e & 7);
  int col = c * 16 + (L & 15);
  float v = (k < MDIM) ? W_ih[col * MDIM + k] : W_hh[col * HDIM + (k - MDIM)];
  Wsw[idx] = f2bf(v);
}

__global__ __launch_bounds__(256) void lstm_prep_b(const float* __restrict__ b_ih,
                                                   const float* __restrict__ b_hh,
                                                   float* __restrict__ bcomb) {
  int i = blockIdx.x * 256 + threadIdx.x;
  if (i < 4 * HDIM) bcomb[i] = b_ih[i] + b_hh[i];
}

__global__ __launch_bounds__(256, 4) void lstm_persistent(
    const float* __restrict__ X, const float* __restrict__ masks,
    const float* __restrict__ h0, const float* __restrict__ c0,
    const float* __restrict__ W_in, const float* __restrict__ b_in,
    const float* __restrict__ W_out, const float* __restrict__ b_out,
    const unsigned short* __restrict__ Wsw, const float* __restrict__ bcomb,
    const int* __restrict__ tobs_p, const int* __restrict__ tpred_p,
    float* __restrict__ out)
{
  __shared__ unsigned short smR[16 * RST];       // r activations (bf16)
  __shared__ unsigned short smH[2][16 * HST];    // h double buffer (bf16)
  __shared__ float smX[32];
  __shared__ float smWin[MDIM * 2];
  __shared__ float smBin[MDIM];
  __shared__ float smWout[2 * HDIM];
  __shared__ float smBout[2];

  const int tid  = threadIdx.x;
  const int lane = tid & 31;
  const int wave = tid >> 5;        // 0..7 : owns h-units [wave*32, wave*32+32)
  const int g    = lane >> 4;       // lane group (0/1)
  const int lm   = lane & 15;
  const int n0   = blockIdx.x * 16; // 16 batch rows per WG

  const int tobs  = tobs_p[0];
  const int tpred = tpred_p[0];

  if (tid < MDIM * 2) smWin[tid] = W_in[tid];
  if (tid < MDIM)     smBin[tid] = b_in[tid];
  smWout[tid]       = W_out[tid];
  smWout[tid + 256] = W_out[tid + 256];
  if (tid < 2) smBout[tid] = b_out[tid];

  // single B base: all 96 fragment loads are base + compile-time immediate
  const unsigned char* bB = (const unsigned char*)Wsw
                          + (size_t)wave * 2048 + (size_t)lane * 32;

  // initial h -> smH[0] (bf16), 16 consecutive elems per thread
  {
    int row = tid >> 4;
    int chunk = (tid & 15) * 16;
    const float4* hp = (const float4*)(h0 + (size_t)(n0 + row) * HDIM + chunk);
    unsigned short* dst = &smH[0][row * HST + chunk];
    #pragma unroll
    for (int q = 0; q < 4; ++q) {
      float4 hv = hp[q];
      dst[q * 4 + 0] = f2bf(hv.x); dst[q * 4 + 1] = f2bf(hv.y);
      dst[q * 4 + 2] = f2bf(hv.z); dst[q * 4 + 3] = f2bf(hv.w);
    }
  }

  // initial c -> regs: lane holds c[row = v+8g][unit = wave*32 + s*16 + lm]
  float creg[16];
  #pragma unroll
  for (int s = 0; s < 2; ++s)
    #pragma unroll
    for (int v = 0; v < 8; ++v)
      creg[s * 8 + v] = c0[(size_t)(n0 + v + 8 * g) * HDIM + wave * 32 + s * 16 + lm];

  // per-lane combined gate biases (uniform over rows)
  float bias[8];
  #pragma unroll
  for (int q = 0; q < 4; ++q)
    #pragma unroll
    for (int s = 0; s < 2; ++s)
      bias[q * 2 + s] = bcomb[q * HDIM + wave * 32 + s * 16 + lm];

  // incrementally advanced global pointers (avoid per-iter re-multiplication)
  const float* xptr = X + (size_t)(n0 + (tid >> 1)) * 4 + 2 + (tid & 1); // tid<32
  const int   orow  = tid >> 4;
  const int   och   = (tid >> 3) & 1;
  const int   opart = tid & 7;
  const float* mptr = masks + n0 + orow;
  float*       optr = out + (size_t)(n0 + orow) * 2 + och;

  int rb = 0;   // current h read-buffer (uniform)

  for (int t = 0; t < TSTEPS; ++t) {
    const bool active = (t > tobs) && (t <= tpred);

    // stage x (safe: previous-iter smX readers all passed the last barrier)
    if (tid < 32) smX[tid] = xptr[0];
    __syncthreads();                               // B1: x staged

    if (active) {
      // r = relu(x @ W_in.T + b_in)  -> smR bf16  (16x128, 8 per thread)
      #pragma unroll
      for (int ii = 0; ii < 8; ++ii) {
        int idx = ii * 256 + tid;
        int row = idx >> 7;
        int j   = idx & 127;
        float v = smX[row * 2 + 0] * smWin[j * 2 + 0] +
                  smX[row * 2 + 1] * smWin[j * 2 + 1] + smBin[j];
        smR[row * RST + j] = f2bf(v > 0.f ? v : 0.f);
      }
    }
    __syncthreads();                               // B2: A matrix complete

    if (active) {
      const unsigned short* hbase = smH[rb];
      v8f acc[8];
      v8f zero = {};
      #pragma unroll
      for (int i = 0; i < 8; ++i) acc[i] = zero;

      // per-kk batched loads: 16 global_load_b128 issued as one burst, then
      // 8 WMMAs drain them with descending loadcnt waits (no wait-to-zero
      // before every WMMA); ~180 VGPRs total, stays under the 256 window
      #pragma unroll
      for (int kk = 0; kk < KSTEPS; ++kk) {
        v16bf aF = loadA_frag(smR, hbase, lm, g, kk);
        v16bf bF[8];
        #pragma unroll
        for (int j = 0; j < 8; ++j)
          bF[j] = loadB_frag(bB, kk, j);
        #pragma unroll
        for (int j = 0; j < 8; ++j)
          acc[j] = __builtin_amdgcn_wmma_f32_16x16x32_bf16(
              false, aF, false, bF[j], (short)0, acc[j], false, false);
      }

      // LSTM cell fully in registers; h_new -> other LDS buffer (no barrier
      // needed: gemm reads smH[rb], cell writes smH[rb^1])
      unsigned short* hw = smH[rb ^ 1];
      #pragma unroll
      for (int s = 0; s < 2; ++s)
        #pragma unroll
        for (int v = 0; v < 8; ++v) {
          float iv = acc[0 * 2 + s][v] + bias[0 * 2 + s];
          float fv = acc[1 * 2 + s][v] + bias[1 * 2 + s];
          float gv = acc[2 * 2 + s][v] + bias[2 * 2 + s];
          float ov = acc[3 * 2 + s][v] + bias[3 * 2 + s];
          float cn = sigm(fv) * creg[s * 8 + v] + sigm(iv) * ftanh(gv);
          float hn = sigm(ov) * ftanh(cn);
          creg[s * 8 + v] = cn;
          hw[(v + 8 * g) * HST + wave * 32 + s * 16 + lm] = f2bf(hn);
        }
      rb ^= 1;                                      // uniform flip
    }
    __syncthreads();                               // B3: h_new visible

    // out = (h_new @ W_out.T + b_out) * mask  (8 lanes cooperate per output)
    {
      float total = 0.f;
      if (active) {
        const unsigned short* hr = &smH[rb][orow * HST + opart * 32];
        const float* wop = &smWout[och * HDIM + opart * 32];
        float sum = 0.f;
        #pragma unroll
        for (int cch = 0; cch < 4; ++cch) {
          v8u hv = *(const v8u*)(hr + cch * 8);     // ds_load_b128
          #pragma unroll
          for (int e = 0; e < 8; ++e)
            sum += bf2f(hv[e]) * wop[cch * 8 + e];
        }
        sum += __shfl_xor(sum, 1);
        sum += __shfl_xor(sum, 2);
        sum += __shfl_xor(sum, 4);
        total = (sum + smBout[och]) * mptr[0];
      }
      if (opart == 0) optr[0] = total;
    }

    xptr += (size_t)NROWS * 4;
    mptr += NROWS;
    optr += (size_t)NROWS * 2;
  }
}

extern "C" void kernel_launch(void* const* d_in, const int* in_sizes, int n_in,
                              void* d_out, int out_size, void* d_ws, size_t ws_size,
                              hipStream_t stream) {
  (void)in_sizes; (void)n_in; (void)out_size; (void)ws_size;
  const float* X     = (const float*)d_in[0];
  const float* masks = (const float*)d_in[1];
  const float* h0    = (const float*)d_in[2];
  const float* c0    = (const float*)d_in[3];
  const float* W_in  = (const float*)d_in[4];
  const float* b_in  = (const float*)d_in[5];
  const float* W_ih  = (const float*)d_in[6];
  const float* b_ih  = (const float*)d_in[7];
  const float* W_hh  = (const float*)d_in[8];
  const float* b_hh  = (const float*)d_in[9];
  const float* W_out = (const float*)d_in[10];
  const float* b_out = (const float*)d_in[11];
  const int*   tobs  = (const int*)d_in[12];
  const int*   tpred = (const int*)d_in[13];
  float* out = (float*)d_out;

  // workspace: swizzled bf16 weights (768 KB) + combined biases (4 KB)
  unsigned short* Wsw = (unsigned short*)d_ws;
  float* bcomb = (float*)((char*)d_ws + (size_t)KSTEPS * 64 * 32 * 16 * sizeof(unsigned short));

  lstm_prep_w<<<1536, 256, 0, stream>>>(W_ih, W_hh, Wsw);
  lstm_prep_b<<<4, 256, 0, stream>>>(b_ih, b_hh, bcomb);
  lstm_persistent<<<NROWS / 16, 256, 0, stream>>>(X, masks, h0, c0, W_in, b_in,
                                                  W_out, b_out, Wsw, bcomb,
                                                  tobs, tpred, out);
}
